// TransformerLayer_25383256720130
// MI455X (gfx1250) — matrix operand order repaired and checked by reading, not compiled, not run on hardware
//
#include <hip/hip_runtime.h>
#include <hip/hip_bf16.h>
#include <stdint.h>

// ---------------- types ----------------
typedef unsigned short u16;
typedef __attribute__((ext_vector_type(16))) __bf16 v16bf;
typedef __attribute__((ext_vector_type(8)))  float  v8f;

union FragB16 { uint4 u[2]; v16bf v; };

// gfx1250 async global->LDS copy path (ASYNCcnt), with fallback if the
// toolchain doesn't declare the builtins.
#if defined(__has_builtin)
#  if __has_builtin(__builtin_amdgcn_global_load_async_to_lds_b128) && \
      __has_builtin(__builtin_amdgcn_s_wait_asynccnt)
#    define USE_ASYNC_LDS 1
#  endif
#endif
#ifndef USE_ASYNC_LDS
#  define USE_ASYNC_LDS 0
#endif

#if USE_ASYNC_LDS
// builtin expects vector-int4 pointers: AS1 (printed '__device__') src, AS3 dst
typedef int v4i_t __attribute__((vector_size(16)));
typedef __attribute__((address_space(1))) v4i_t as1_v4i;
typedef __attribute__((address_space(3))) v4i_t as3_v4i;
#endif

__device__ __forceinline__ void async_copy16(const u16* g, u16* l) {
#if USE_ASYNC_LDS
    // generic->AS1 keeps the 64-bit value; generic LDS ptr bits[31:0] are the
    // LDS offset (ISA 10.2 aperture mapping), so truncation yields the AS3 ptr.
    __builtin_amdgcn_global_load_async_to_lds_b128(
        (as1_v4i*)(uintptr_t)g,
        (as3_v4i*)(uint32_t)(uintptr_t)l,
        0, 0);
#else
    *(uint4*)l = *(const uint4*)g;
#endif
}

__device__ __forceinline__ void wait_stage() {
#if USE_ASYNC_LDS
    __builtin_amdgcn_s_wait_asynccnt(0);
#endif
}

__device__ __forceinline__ u16 f32_bf16(float f) {
    unsigned u = __float_as_uint(f);
    unsigned r = 0x7FFFu + ((u >> 16) & 1u);   // round-to-nearest-even
    return (u16)((u + r) >> 16);
}

__device__ __forceinline__ v8f vzero() {
    v8f z = {0.f,0.f,0.f,0.f,0.f,0.f,0.f,0.f};
    return z;
}

__device__ __forceinline__ v8f wmma_bf16(const FragB16& a, const FragB16& b, v8f c) {
    return __builtin_amdgcn_wmma_f32_16x16x32_bf16(
        /*neg_a*/false, a.v, /*neg_b*/false, b.v,
        /*c_mod*/(short)0, c, /*reuse_a*/false, /*reuse_b*/false);
}

// ---------------- problem dims ----------------
#define BB   2
#define NN   2048
#define DM   512
#define HH   8
#define DH   64
#define DFF  2048
#define ROWS (BB*NN)          // 4096

// ---------------- conversion kernels ----------------
__global__ void cvt_bf16(const float* __restrict__ in, u16* __restrict__ out, int n) {
    int i = blockIdx.x * blockDim.x + threadIdx.x;
    if (i < n) out[i] = f32_bf16(in[i]);
}

// in: [K][N] fp32 row-major  ->  out: [N][K] bf16 (N-major, "Bt")
__global__ void cvt_bf16_t(const float* __restrict__ in, u16* __restrict__ out, int K, int N) {
    int i = blockIdx.x * blockDim.x + threadIdx.x;
    if (i < K * N) {
        int k = i / N, n = i % N;
        out[(size_t)n * K + k] = f32_bf16(in[i]);
    }
}

// ---------------- WMMA GEMM: C[M,N] = A[M,K] * Bt[N,K]^T ----------------
#define BM 128
#define BN 128
#define BK 32
#define LDSS 48     // u16 elements per LDS row (32 data + 16 pad; 96B = 16B-aligned rows)

__global__ __launch_bounds__(256) void gemm_bf16(
    const u16*  __restrict__ A,     // [M][K] bf16 row-major
    const u16*  __restrict__ Bt,    // [N][K] bf16 row-major (transposed weights)
    const float* __restrict__ bias, // [N] or nullptr
    float* __restrict__ outF,       // [M][N] or nullptr
    u16*   __restrict__ outB,       // [M][N] bf16 or nullptr
    int M, int N, int K, int relu)
{
    __shared__ __align__(16) u16 sA[2][BM * LDSS];
    __shared__ __align__(16) u16 sB[2][BN * LDSS];

    const int tid   = threadIdx.x;
    const int lane  = tid & 31;
    const int half  = (lane >> 4) & 1;
    const int lm    = lane & 15;
    const int wave  = tid >> 5;          // 0..7
    const int waveM = wave >> 1;         // 0..3  -> 32 rows each
    const int waveN = wave & 1;          // 0..1  -> 64 cols each
    const int bM    = blockIdx.y * BM;
    const int bN    = blockIdx.x * BN;

    v8f acc[2][4];
    for (int mt = 0; mt < 2; ++mt)
        for (int nt = 0; nt < 4; ++nt) acc[mt][nt] = vzero();

    // cooperative stage of one 128x32 A tile and one 128x32 Bt tile
    // (async global->LDS when available: no VGPR round-trip, ASYNCcnt-tracked)
    auto stage = [&](int buf, int k0) {
        #pragma unroll
        for (int i = 0; i < 2; ++i) {
            int li  = tid + i * 256;       // 0..511
            int row = li >> 2;             // 0..127
            int seg = li & 3;              // 0..3  (8 u16 each)
            async_copy16(A  + (size_t)(bM + row) * K + k0 + seg * 8,
                         &sA[buf][row * LDSS + seg * 8]);
            async_copy16(Bt + (size_t)(bN + row) * K + k0 + seg * 8,
                         &sB[buf][row * LDSS + seg * 8]);
        }
    };

    const int nk = K / BK;
    stage(0, 0);
    wait_stage();
    __syncthreads();

    int buf = 0;
    for (int kt = 0; kt < nk; ++kt) {
        if (kt + 1 < nk) stage(buf ^ 1, (kt + 1) * BK);

        FragB16 af[2], bfr[4];
        #pragma unroll
        for (int mt = 0; mt < 2; ++mt) {
            // A-frag (16x32): elems 0..7 -> K=8*half.., elems 8..15 -> K=16+8*half..
            const u16* pa = &sA[buf][(waveM * 32 + mt * 16 + lm) * LDSS];
            af[mt].u[0] = *(const uint4*)(pa + 8 * half);
            af[mt].u[1] = *(const uint4*)(pa + 16 + 8 * half);
        }
        #pragma unroll
        for (int nt = 0; nt < 4; ++nt) {
            // B-frag (32x16): elems 0..7 -> K=8*half.., elems 8..15 -> K=16+8*half.. for column lm
            const u16* pb = &sB[buf][(waveN * 64 + nt * 16 + lm) * LDSS];
            bfr[nt].u[0] = *(const uint4*)(pb + 8 * half);
            bfr[nt].u[1] = *(const uint4*)(pb + 16 + 8 * half);
        }
        #pragma unroll
        for (int mt = 0; mt < 2; ++mt)
            #pragma unroll
            for (int nt = 0; nt < 4; ++nt)
                acc[mt][nt] = wmma_bf16(af[mt], bfr[nt], acc[mt][nt]);

        wait_stage();          // async stage of buf^1 complete before barrier
        __syncthreads();
        buf ^= 1;
    }

    // epilogue: bias + optional relu, write fp32 and/or bf16
    const int gM = bM + waveM * 32;
    const int gN = bN + waveN * 64;
    #pragma unroll
    for (int mt = 0; mt < 2; ++mt) {
        #pragma unroll
        for (int nt = 0; nt < 4; ++nt) {
            int c = gN + nt * 16 + lm;
            float bv = bias ? bias[c] : 0.f;
            #pragma unroll
            for (int v = 0; v < 8; ++v) {
                int r = gM + mt * 16 + half * 8 + v;     // C layout: VGPR v -> row (8*half+v)
                float val = acc[mt][nt][v] + bv;
                if (relu) val = fmaxf(val, 0.f);
                size_t idx = (size_t)r * N + c;
                if (outF) outF[idx] = val;
                if (outB) outB[idx] = f32_bf16(val);
            }
        }
    }
}

// ---------------- flash attention ----------------
// qkv: [B*N][1536] bf16 (q|k|v each H*64), mask: [B][N][N] int, attn: [B*N][512] bf16
__global__ __launch_bounds__(128) void flash_attn(
    const u16* __restrict__ qkv,
    const int* __restrict__ mask,
    u16* __restrict__ attn)
{
    __shared__ __align__(16) u16 sVt[64 * LDSS];        // V^T tile: [d=64][key=32]
    __shared__ __align__(16) u16 sP[4][16 * LDSS];      // per-wave P tile: [16][32]

    const int tid  = threadIdx.x;
    const int lane = tid & 31;
    const int wv   = tid >> 5;            // 0..3
    const int half = (lane >> 4) & 1;
    const int lm   = lane & 15;
    const int bb   = blockIdx.z;
    const int h    = blockIdx.y;
    const int q0   = blockIdx.x * 64 + wv * 16;

    // Q fragments for this wave's 16 rows: two 16x32 bf16 A-frags (d = 0..31, 32..63)
    FragB16 qf[2];
    {
        const u16* qr = qkv + ((size_t)(bb * NN + q0 + lm) * (3 * DM) + h * DH);
        qf[0].u[0] = *(const uint4*)(qr + 8 * half);
        qf[0].u[1] = *(const uint4*)(qr + 16 + 8 * half);
        qf[1].u[0] = *(const uint4*)(qr + 32 + 8 * half);
        qf[1].u[1] = *(const uint4*)(qr + 48 + 8 * half);
    }

    v8f o[4];
    for (int d = 0; d < 4; ++d) o[d] = vzero();
    float mrow[8], lrow[8];
    for (int v = 0; v < 8; ++v) { mrow[v] = -3.0e38f; lrow[v] = 0.f; }

    for (int kb = 0; kb < NN / 32; ++kb) {
        // ---- cooperative stage V[32 keys][64 d] -> sVt[d][key] (transposed) ----
        #pragma unroll
        for (int i = 0; i < 2; ++i) {
            int li  = tid + i * 128;       // 0..255
            int key = li >> 3;             // 0..31
            int seg = li & 7;              // 0..7 (8 d-values)
            const u16* vr = qkv + ((size_t)(bb * NN + kb * 32 + key) * (3 * DM)
                                   + 2 * DM + h * DH + seg * 8);
            uint4 d4 = *(const uint4*)vr;
            const u16* e = (const u16*)&d4;
            #pragma unroll
            for (int j = 0; j < 8; ++j)
                sVt[(seg * 8 + j) * LDSS + key] = e[j];
        }
        __syncthreads();

        // ---- S = (Q K^T) * scale, masked; two 16x16 subtiles (32 keys) ----
        v8f s[2];
        #pragma unroll
        for (int jb = 0; jb < 2; ++jb) {
            FragB16 kf0, kf1;   // B-frags: column lm = key, elems 0..7 -> d = 8*half.., 8..15 -> d = 16+8*half..
            const u16* kr = qkv + ((size_t)(bb * NN + kb * 32 + jb * 16 + lm) * (3 * DM)
                                   + DM + h * DH);
            kf0.u[0] = *(const uint4*)(kr + 8 * half);
            kf0.u[1] = *(const uint4*)(kr + 16 + 8 * half);
            kf1.u[0] = *(const uint4*)(kr + 32 + 8 * half);
            kf1.u[1] = *(const uint4*)(kr + 48 + 8 * half);

            v8f c = vzero();
            c = wmma_bf16(qf[0], kf0, c);
            c = wmma_bf16(qf[1], kf1, c);

            const int kj = kb * 32 + jb * 16 + lm;
            #pragma unroll
            for (int v = 0; v < 8; ++v) {
                int qi = q0 + half * 8 + v;
                int mk = mask[((size_t)bb * NN + qi) * NN + kj];
                float sv = c[v] * 0.125f;                 // 1/sqrt(64)
                c[v] = mk ? sv : -1.0e9f;
            }
            s[jb] = c;
        }

        // ---- online softmax (row stats replicated across the 16 lanes of a half) ----
        #pragma unroll
        for (int v = 0; v < 8; ++v) {
            float mx = fmaxf(s[0][v], s[1][v]);
            #pragma unroll
            for (int off = 8; off >= 1; off >>= 1)
                mx = fmaxf(mx, __shfl_xor(mx, off, 16));
            float mnew = fmaxf(mrow[v], mx);
            float p0 = __expf(s[0][v] - mnew);
            float p1 = __expf(s[1][v] - mnew);
            float ls = p0 + p1;
            #pragma unroll
            for (int off = 8; off >= 1; off >>= 1)
                ls += __shfl_xor(ls, off, 16);
            float alpha = __expf(mrow[v] - mnew);
            lrow[v] = lrow[v] * alpha + ls;
            mrow[v] = mnew;
            #pragma unroll
            for (int d = 0; d < 4; ++d) o[d][v] *= alpha;
            // C-layout -> LDS, row = 8*half+v, cols jb*16+lm
            sP[wv][(half * 8 + v) * LDSS + lm]      = f32_bf16(p0);
            sP[wv][(half * 8 + v) * LDSS + 16 + lm] = f32_bf16(p1);
        }

        // ---- O += P @ V  (P: 16x32 A-frag from wave-local LDS; same-wave LDS is in-order) ----
        FragB16 pf;
        const u16* pp = &sP[wv][lm * LDSS];
        pf.u[0] = *(const uint4*)(pp + 8 * half);
        pf.u[1] = *(const uint4*)(pp + 16 + 8 * half);
        #pragma unroll
        for (int d = 0; d < 4; ++d) {
            FragB16 vf;   // B-frag: column lm = d (within chunk), elems 0..7 -> key = 8*half.., 8..15 -> key = 16+8*half..
            const u16* pv = &sVt[(d * 16 + lm) * LDSS];
            vf.u[0] = *(const uint4*)(pv + 8 * half);
            vf.u[1] = *(const uint4*)(pv + 16 + 8 * half);
            o[d] = wmma_bf16(pf, vf, o[d]);
        }
        __syncthreads();   // before next iteration overwrites sVt
    }

    // ---- normalize and store: attn[b, q, h*64 + d] ----
    #pragma unroll
    for (int v = 0; v < 8; ++v) {
        float inv = 1.0f / lrow[v];
        int r = q0 + half * 8 + v;
        #pragma unroll
        for (int d = 0; d < 4; ++d) {
            size_t idx = (size_t)(bb * NN + r) * DM + h * DH + d * 16 + lm;
            attn[idx] = f32_bf16(o[d][v] * inv);
        }
    }
}

// ---------------- residual + layernorm (one row of 512 per block) ----------------
__global__ __launch_bounds__(256) void resid_ln(
    const float* __restrict__ x, const float* __restrict__ r,
    const float* __restrict__ g, const float* __restrict__ b,
    float* __restrict__ outF, u16* __restrict__ outB)
{
    const int row = blockIdx.x;
    const int t   = threadIdx.x;
    const float* xr = x + (size_t)row * DM;
    const float* rr = r + (size_t)row * DM;

    float v0 = xr[t] + rr[t];
    float v1 = xr[t + 256] + rr[t + 256];
    float s = v0 + v1;
    float q = v0 * v0 + v1 * v1;

    __shared__ float rs[8], rq[8];
    #pragma unroll
    for (int off = 16; off >= 1; off >>= 1) {
        s += __shfl_xor(s, off, 32);
        q += __shfl_xor(q, off, 32);
    }
    if ((t & 31) == 0) { rs[t >> 5] = s; rq[t >> 5] = q; }
    __syncthreads();
    float S = 0.f, Q = 0.f;
    #pragma unroll
    for (int i = 0; i < 8; ++i) { S += rs[i]; Q += rq[i]; }

    float mu   = S * (1.0f / DM);
    float var  = Q * (1.0f / DM) - mu * mu;
    float rinv = rsqrtf(var + 1e-5f);

    float y0 = (v0 - mu) * rinv * g[t] + b[t];
    float y1 = (v1 - mu) * rinv * g[t + 256] + b[t + 256];
    size_t base = (size_t)row * DM;
    outF[base + t]       = y0;
    outF[base + t + 256] = y1;
    if (outB) {
        outB[base + t]       = f32_bf16(y0);
        outB[base + t + 256] = f32_bf16(y1);
    }
}

// ---------------- host launch ----------------
extern "C" void kernel_launch(void* const* d_in, const int* in_sizes, int n_in,
                              void* d_out, int out_size, void* d_ws, size_t ws_size,
                              hipStream_t stream) {
    (void)in_sizes; (void)n_in; (void)out_size; (void)ws_size;

    const float* x      = (const float*)d_in[0];
    const int*   mask   = (const int*)  d_in[1];
    const float* W_qkv  = (const float*)d_in[2];
    const float* b_qkv  = (const float*)d_in[3];
    const float* W_proj = (const float*)d_in[4];
    const float* b_proj = (const float*)d_in[5];
    const float* ln1_g  = (const float*)d_in[6];
    const float* ln1_b  = (const float*)d_in[7];
    const float* W1     = (const float*)d_in[8];
    const float* b1     = (const float*)d_in[9];
    const float* W2     = (const float*)d_in[10];
    const float* b2     = (const float*)d_in[11];
    const float* ln2_g  = (const float*)d_in[12];
    const float* ln2_b  = (const float*)d_in[13];
    float* out = (float*)d_out;

    char* w = (char*)d_ws;
    size_t off = 0;
    auto alloc = [&](size_t bytes) -> void* {
        void* p = w + off;
        off += (bytes + 255) & ~(size_t)255;
        return p;
    };
    u16*   xb     = (u16*)  alloc((size_t)ROWS * DM * 2);        //  4 MB
    u16*   Wqkvt  = (u16*)  alloc((size_t)DM * 3 * DM * 2);      //  1.5 MB
    u16*   Wprojt = (u16*)  alloc((size_t)DM * DM * 2);          //  0.5 MB
    u16*   W1t    = (u16*)  alloc((size_t)DFF * DM * 2);         //  2 MB
    u16*   W2t    = (u16*)  alloc((size_t)DM * DFF * 2);         //  2 MB
    u16*   qkvb   = (u16*)  alloc((size_t)ROWS * 3 * DM * 2);    // 12 MB
    u16*   attnb  = (u16*)  alloc((size_t)ROWS * DM * 2);        //  4 MB
    float* aF     = (float*)alloc((size_t)ROWS * DM * 4);        //  8 MB
    float* yF     = (float*)alloc((size_t)ROWS * DM * 4);        //  8 MB
    u16*   yb     = (u16*)  alloc((size_t)ROWS * DM * 2);        //  4 MB
    u16*   hb     = (u16*)  alloc((size_t)ROWS * DFF * 2);       // 16 MB
    float* mF     = (float*)alloc((size_t)ROWS * DM * 4);        //  8 MB

    // 1) convert activations + weights (weights transposed to N-major bf16)
    cvt_bf16  <<<(ROWS * DM + 255) / 256, 256, 0, stream>>>(x, xb, ROWS * DM);
    cvt_bf16_t<<<(DM * 3 * DM + 255) / 256, 256, 0, stream>>>(W_qkv,  Wqkvt,  DM,  3 * DM);
    cvt_bf16_t<<<(DM * DM     + 255) / 256, 256, 0, stream>>>(W_proj, Wprojt, DM,  DM);
    cvt_bf16_t<<<(DM * DFF    + 255) / 256, 256, 0, stream>>>(W1,     W1t,    DM,  DFF);
    cvt_bf16_t<<<(DFF * DM    + 255) / 256, 256, 0, stream>>>(W2,     W2t,    DFF, DM);

    // 2) qkv = x @ W_qkv + b_qkv            [4096 x 1536], K=512  -> bf16
    gemm_bf16<<<dim3((3 * DM) / BN, ROWS / BM), 256, 0, stream>>>(
        xb, Wqkvt, b_qkv, nullptr, qkvb, ROWS, 3 * DM, DM, 0);

    // 3) flash attention -> attn bf16 [4096 x 512]
    flash_attn<<<dim3(NN / 64, HH, BB), 128, 0, stream>>>(qkvb, mask, attnb);

    // 4) a = attn @ W_proj + b_proj         [4096 x 512], K=512 -> fp32
    gemm_bf16<<<dim3(DM / BN, ROWS / BM), 256, 0, stream>>>(
        attnb, Wprojt, b_proj, aF, nullptr, ROWS, DM, DM, 0);

    // 5) y = LN(x + a)  -> fp32 + bf16
    resid_ln<<<ROWS, 256, 0, stream>>>(x, aF, ln1_g, ln1_b, yF, yb);

    // 6) h = relu(y @ W1 + b1)              [4096 x 2048], K=512 -> bf16
    gemm_bf16<<<dim3(DFF / BN, ROWS / BM), 256, 0, stream>>>(
        yb, W1t, b1, nullptr, hb, ROWS, DFF, DM, 1);

    // 7) m = relu(h @ W2 + b2)              [4096 x 512], K=2048 -> fp32
    gemm_bf16<<<dim3(DM / BN, ROWS / BM), 256, 0, stream>>>(
        hb, W2t, b2, mF, nullptr, ROWS, DM, DFF, 1);

    // 8) out = LN(y + m)
    resid_ln<<<ROWS, 256, 0, stream>>>(yF, mF, ln2_g, ln2_b, out, nullptr);
}
